// SpectralFeatureExtractor_47193100648813
// MI455X (gfx1250) — matrix operand order repaired and checked by reading, not compiled, not run on hardware
//
#include <hip/hip_runtime.h>

#define BATCH   2097152
#define GRID    1024
#define BLOCK   256

// d_ws layout (floats)
#define WEFF 0      // 64x32 fused comb weight
#define BEFF 2048   // 64   fused comb bias
#define WBAR 2112   // 32   column sums of W_eff
#define BBAR 2144   // 1    sum of bias_eff

typedef _Float16 v16h __attribute__((ext_vector_type(16)));
typedef float    v8f  __attribute__((ext_vector_type(8)));

union AFrag { v16h h; uint4 u4[2]; };
union H2U  { _Float16 h[2]; unsigned int u; };

__device__ __forceinline__ void dsfence() {
  // intra-wave LDS ordering: wait all DS ops + compiler barrier
  asm volatile("s_wait_dscnt 0" ::: "memory");
}

// Branch-free tanh: 1 - 2/(exp2(2*log2(e)*x)+1); saturates to +/-1, NaN propagates
__device__ __forceinline__ float fast_tanh(float x) {
  float e = __builtin_amdgcn_exp2f(x * 2.8853900817779268f);
  return 1.0f - 2.0f * __builtin_amdgcn_rcpf(e + 1.0f);
}

// A-fragment (16-bit 16x32): lane<16 -> K{0..7,16..23}, lane>=16 -> K{8..15,24..31}
__device__ __forceinline__ v16h load_afrag(const _Float16* rowp, int kh) {
  AFrag f;
  f.u4[0] = *reinterpret_cast<const uint4*>(rowp + 8 * kh);
  f.u4[1] = *reinterpret_cast<const uint4*>(rowp + 16 + 8 * kh);
  return f.h;
}

// B-fragment (16-bit 32x16): lane = column N, lanes<16 K{0..7,16..23}, lanes>=16 K{8..15,24..31}
// wrow points at W[n][k0] (row-major weight, pre-offset by K chunk)
__device__ __forceinline__ v16h make_bfrag(const float* wrow, int g) {
  v16h r;
#pragma unroll
  for (int e = 0; e < 16; ++e) r[e] = (_Float16)wrow[8 * g + (e & 7) + 16 * (e >> 3)];
  return r;
}

// ---- Prep kernel: fold branch layer-2s into the comb GEMM; one 64-thread block ----
// W_eff = W_comb @ blockdiag(W_rgb2, W_ir2, W_n2); bias_eff = W_comb @ b2 + b_comb
// wbar[k] = sum_n W_eff[n][k]; bbar = sum_n bias_eff[n]  (for the linear mean trick)
__global__ void prep_kernel(
    const float* __restrict__ w_rgb2, const float* __restrict__ b_rgb2,
    const float* __restrict__ w_ir2,  const float* __restrict__ b_ir2,
    const float* __restrict__ w_n2,   const float* __restrict__ b_n2,
    const float* __restrict__ w_comb, const float* __restrict__ b_comb,
    float* __restrict__ ws) {
  __shared__ float sWeff[64 * 32];
  __shared__ float sBias[64];
  const int n = threadIdx.x;
  if (n < 64) {
    const float* wcr = w_comb + n * 32;
    float be = b_comb[n];
    for (int j = 0; j < 16; ++j) be += wcr[j] * b_rgb2[j];
    for (int j = 0; j < 8; ++j)  be += wcr[16 + j] * b_ir2[j];
    for (int j = 0; j < 8; ++j)  be += wcr[24 + j] * b_n2[j];
    sBias[n] = be;
    ws[BEFF + n] = be;
    for (int k = 0; k < 16; ++k) {
      float a = 0.f;
      for (int j = 0; j < 16; ++j) a += wcr[j] * w_rgb2[j * 16 + k];
      sWeff[n * 32 + k] = a;
    }
    for (int k = 0; k < 8; ++k) {
      float a = 0.f;
      for (int j = 0; j < 8; ++j) a += wcr[16 + j] * w_ir2[j * 8 + k];
      sWeff[n * 32 + 16 + k] = a;
    }
    for (int k = 0; k < 8; ++k) {
      float a = 0.f;
      for (int j = 0; j < 8; ++j) a += wcr[24 + j] * w_n2[j * 8 + k];
      sWeff[n * 32 + 24 + k] = a;
    }
    for (int k = 0; k < 32; ++k) ws[WEFF + n * 32 + k] = sWeff[n * 32 + k];
  }
  __syncthreads();
  if (n < 32) {
    float a = 0.f;
    for (int m = 0; m < 64; ++m) a += sWeff[m * 32 + n];
    ws[WBAR + n] = a;
  }
  if (n == 32) {
    float a = 0.f;
    for (int m = 0; m < 64; ++m) a += sBias[m];
    ws[BBAR] = a;
  }
}

__global__ __launch_bounds__(BLOCK) void spectral_kernel(
    const float* __restrict__ spectral,
    const float* __restrict__ w_rgb1, const float* __restrict__ b_rgb1,
    const float* __restrict__ w_ir1,  const float* __restrict__ b_ir1,
    const float* __restrict__ w_n1,   const float* __restrict__ b_n1,
    const float* __restrict__ ln_g,   const float* __restrict__ ln_b,
    const float* __restrict__ w_s1,   const float* __restrict__ b_s1,
    const float* __restrict__ w_s2,   const float* __restrict__ b_s2,
    const float* __restrict__ fused,
    float* __restrict__ out) {

  __shared__ _Float16 sAc[8][32 * 32];    // per-wave relu_h (K=32 A matrix)
  __shared__ _Float16 sBase[8][16 * 64];  // per-wave, per-mtile base (K=64 A matrix)
  __shared__ _Float16 sS1[8][16 * 32];    // per-wave, per-mtile relu(s1) (K=32 A matrix)
  __shared__ __align__(16) float sMu[8][32];  // per-wave row means
  __shared__ __align__(16) float sQ[8][16];   // per-wave per-mtile sum of squares

  const int tid  = threadIdx.x;
  const int lane = tid & 31;
  const int wv   = tid >> 5;
  const int nl   = lane & 15;
  const int kh   = lane >> 4;  // K-half selector for A and B fragments

  // B fragments (loop-invariant, live in VGPRs); comb weights come pre-fused from ws
  v16h Bc[4], Bs1f[2][2], Bs2f[4];
  float biasC[4];
#pragma unroll
  for (int t = 0; t < 4; ++t) {
    int n = nl + 16 * t;
    Bc[t] = make_bfrag(fused + WEFF + n * 32, kh);
    biasC[t] = fused[BEFF + n];
  }
#pragma unroll
  for (int t = 0; t < 2; ++t)
#pragma unroll
    for (int kc = 0; kc < 2; ++kc)
      Bs1f[t][kc] = make_bfrag(w_s1 + (nl + 16 * t) * 64 + 32 * kc, kh);
#pragma unroll
  for (int t = 0; t < 4; ++t) Bs2f[t] = make_bfrag(w_s2 + (nl + 16 * t) * 32, kh);

  // lane-dependent loop-invariant params
  float lnG[4], lnB[4], biasS1[2], biasS2[4];
#pragma unroll
  for (int t = 0; t < 4; ++t) {
    int n = nl + 16 * t;
    lnG[t] = ln_g[n]; lnB[t] = ln_b[n]; biasS2[t] = b_s2[n];
  }
#pragma unroll
  for (int t = 0; t < 2; ++t) biasS1[t] = b_s1[nl + 16 * t];

#pragma unroll 1
  for (int rb = blockIdx.x * 256; rb < BATCH; rb += GRID * 256) {
    const int waveRow = rb + wv * 32;

    // ---- Stage 1: per-lane first-layer MLPs (88 FMAs/row) + linear row-mean ----
    const float* sp = spectral + (size_t)(waveRow + lane) * 8;
    float4 p0 = reinterpret_cast<const float4*>(sp)[0];
    float4 p1 = reinterpret_cast<const float4*>(sp)[1];
    float x0 = p0.x, x1 = p0.y, x2 = p0.z, x3 = p0.w;
    float x4 = p1.x, x5 = p1.y, x6 = p1.z, x7 = p1.w;

    float cmb[32];
#pragma unroll
    for (int j = 0; j < 16; ++j)
      cmb[j] = fmaxf(b_rgb1[j] + w_rgb1[j * 3 + 0] * x0 + w_rgb1[j * 3 + 1] * x1 +
                     w_rgb1[j * 3 + 2] * x2, 0.f);
#pragma unroll
    for (int j = 0; j < 8; ++j)
      cmb[16 + j] = fmaxf(b_ir1[j] + w_ir1[j * 2 + 0] * x3 + w_ir1[j * 2 + 1] * x4, 0.f);
#pragma unroll
    for (int j = 0; j < 8; ++j)
      cmb[24 + j] = fmaxf(b_n1[j] + w_n1[j * 3 + 0] * x5 + w_n1[j * 3 + 1] * x6 +
                          w_n1[j * 3 + 2] * x7, 0.f);

    // row mean of h via linearity: mu = (cmb . wbar + bbar)/64  (wbar uniform -> s_loads)
    {
      float hsum = fused[BBAR];
#pragma unroll
      for (int k = 0; k < 32; ++k) hsum = fmaf(fused[WBAR + k], cmb[k], hsum);
      sMu[wv][lane] = hsum * 0.015625f;
    }

    // pack relu_h -> f16, row-major stride 32, per-wave LDS tile
    {
      unsigned int* dst = reinterpret_cast<unsigned int*>(&sAc[wv][lane * 32]);
#pragma unroll
      for (int d = 0; d < 16; ++d) {
        H2U p;
        p.h[0] = (_Float16)cmb[2 * d];
        p.h[1] = (_Float16)cmb[2 * d + 1];
        dst[d] = p.u;
      }
    }
    dsfence();

    // ---- Per-M-tile pipeline: (fused) comb GEMM -> LN/tanh -> s1 -> s2 ----
#pragma unroll
    for (int mt = 0; mt < 2; ++mt) {
      // zero the per-row sum-of-squares accumulator (in-order DS: safe without fence)
      if (lane < 16) sQ[wv][lane] = 0.f;

      // fused comb GEMM (16x64 tile, K=32): 4 WMMAs
      v16h aC = load_afrag(&sAc[wv][(16 * mt + nl) * 32], kh);
      v8f hacc[4];
#pragma unroll
      for (int nt = 0; nt < 4; ++nt) {
        v8f c;
#pragma unroll
        for (int r = 0; r < 8; ++r) c[r] = biasC[nt];
        hacc[nt] = __builtin_amdgcn_wmma_f32_16x16x32_f16(
            false, aC, false, Bc[nt], (short)0, c, false, false);
      }

      // accumulate per-row sum of squares with LDS f32 atomics (16 lanes/row)
#pragma unroll
      for (int r = 0; r < 8; ++r) {
        float q = hacc[0][r] * hacc[0][r];
        q = fmaf(hacc[1][r], hacc[1][r], q);
        q = fmaf(hacc[2][r], hacc[2][r], q);
        q = fmaf(hacc[3][r], hacc[3][r], q);
        atomicAdd(&sQ[wv][8 * kh + r], q);
      }
      dsfence();

      float muA[8], qA[8];
      *reinterpret_cast<float4*>(&muA[0]) =
          *reinterpret_cast<const float4*>(&sMu[wv][16 * mt + 8 * kh]);
      *reinterpret_cast<float4*>(&muA[4]) =
          *reinterpret_cast<const float4*>(&sMu[wv][16 * mt + 8 * kh + 4]);
      *reinterpret_cast<float4*>(&qA[0]) =
          *reinterpret_cast<const float4*>(&sQ[wv][8 * kh]);
      *reinterpret_cast<float4*>(&qA[4]) =
          *reinterpret_cast<const float4*>(&sQ[wv][8 * kh + 4]);

      // LayerNorm + tanh in C-layout; keep base in regs, write f16 A matrix
      float base[4][8];
#pragma unroll
      for (int r = 0; r < 8; ++r) {
        float mu   = muA[r];
        float var  = qA[r] * 0.015625f - mu * mu;
        float rstd = rsqrtf(var + 1e-5f);
        float murs = mu * rstd;
        int   ml   = 8 * kh + r;  // row within this mtile
#pragma unroll
        for (int nt = 0; nt < 4; ++nt) {
          float z = hacc[nt][r] * rstd - murs;
          float v = fast_tanh(z * lnG[nt] + lnB[nt]);
          base[nt][r] = v;
          sBase[wv][ml * 64 + nl + 16 * nt] = (_Float16)v;
        }
      }
      dsfence();

      // s1 GEMM (K=64): 4 WMMAs (2 N-tiles x 2 accumulating K-chunks)
      v16h aB0 = load_afrag(&sBase[wv][nl * 64 + 0], kh);
      v16h aB1 = load_afrag(&sBase[wv][nl * 64 + 32], kh);
      v8f s1acc[2];
#pragma unroll
      for (int nt = 0; nt < 2; ++nt) {
        v8f c;
#pragma unroll
        for (int r = 0; r < 8; ++r) c[r] = biasS1[nt];
        c = __builtin_amdgcn_wmma_f32_16x16x32_f16(
            false, aB0, false, Bs1f[nt][0], (short)0, c, false, false);
        s1acc[nt] = __builtin_amdgcn_wmma_f32_16x16x32_f16(
            false, aB1, false, Bs1f[nt][1], (short)0, c, false, false);
      }
#pragma unroll
      for (int nt = 0; nt < 2; ++nt)
#pragma unroll
        for (int r = 0; r < 8; ++r)
          sS1[wv][(8 * kh + r) * 32 + nl + 16 * nt] =
              (_Float16)fmaxf(s1acc[nt][r], 0.f);
      dsfence();

      // s2 GEMM (K=32): 4 WMMAs; C initialized with bias + base (folds epilogue)
      v16h aS = load_afrag(&sS1[wv][nl * 32], kh);
#pragma unroll
      for (int nt = 0; nt < 4; ++nt) {
        v8f c;
#pragma unroll
        for (int r = 0; r < 8; ++r) c[r] = biasS2[nt] + base[nt][r];
        c = __builtin_amdgcn_wmma_f32_16x16x32_f16(
            false, aS, false, Bs2f[nt], (short)0, c, false, false);
#pragma unroll
        for (int r = 0; r < 8; ++r) {
          int row = waveRow + 16 * mt + 8 * kh + r;
          out[(size_t)row * 64 + nl + 16 * nt] = c[r];
        }
      }
    }
  }
}

extern "C" void kernel_launch(void* const* d_in, const int* in_sizes, int n_in,
                              void* d_out, int out_size, void* d_ws, size_t ws_size,
                              hipStream_t stream) {
  (void)in_sizes; (void)n_in; (void)out_size; (void)ws_size;
  const float* spectral = (const float*)d_in[0];
  const float* w_rgb1 = (const float*)d_in[1];
  const float* b_rgb1 = (const float*)d_in[2];
  const float* w_rgb2 = (const float*)d_in[3];
  const float* b_rgb2 = (const float*)d_in[4];
  const float* w_ir1  = (const float*)d_in[5];
  const float* b_ir1  = (const float*)d_in[6];
  const float* w_ir2  = (const float*)d_in[7];
  const float* b_ir2  = (const float*)d_in[8];
  const float* w_n1   = (const float*)d_in[9];
  const float* b_n1   = (const float*)d_in[10];
  const float* w_n2   = (const float*)d_in[11];
  const float* b_n2   = (const float*)d_in[12];
  const float* w_comb = (const float*)d_in[13];
  const float* b_comb = (const float*)d_in[14];
  const float* ln_g   = (const float*)d_in[15];
  const float* ln_b   = (const float*)d_in[16];
  const float* w_s1   = (const float*)d_in[17];
  const float* b_s1   = (const float*)d_in[18];
  const float* w_s2   = (const float*)d_in[19];
  const float* b_s2   = (const float*)d_in[20];
  float* out = (float*)d_out;
  float* ws  = (float*)d_ws;

  prep_kernel<<<1, 64, 0, stream>>>(w_rgb2, b_rgb2, w_ir2, b_ir2, w_n2, b_n2,
                                    w_comb, b_comb, ws);
  spectral_kernel<<<GRID, BLOCK, 0, stream>>>(
      spectral, w_rgb1, b_rgb1, w_ir1, b_ir1, w_n1, b_n1, ln_g, ln_b,
      w_s1, b_s1, w_s2, b_s2, ws, out);
}
